// skeleton_lstm_58798102282618
// MI455X (gfx1250) — compile-verified, matrix operand order checked by reading, not run on hardware
//
#include <hip/hip_runtime.h>

typedef _Float16 half_t;
typedef __attribute__((ext_vector_type(16))) _Float16 v16h;
typedef __attribute__((ext_vector_type(8)))  _Float16 v8h;
typedef __attribute__((ext_vector_type(8)))  float    v8f;
typedef __attribute__((ext_vector_type(4)))  int      v4i;

union HFrag { v16h v; v8h h[2]; };

#if defined(__AMDGCN__) && __has_builtin(__builtin_amdgcn_global_load_async_to_lds_b128) && \
    __has_builtin(__builtin_amdgcn_s_wait_asynccnt)
#define HAS_ASYNC_LDS 1
#else
#define HAS_ASYNC_LDS 0
#endif

#if HAS_ASYNC_LDS
typedef __attribute__((address_space(1))) v4i gas_v4i;
typedef __attribute__((address_space(3))) v4i las_v4i;
#endif

// 16-byte global -> LDS copy; async DMA (ASYNCcnt) when available.
__device__ __forceinline__ void cp16_g2l(void* ldst, const void* gsrc) {
#if HAS_ASYNC_LDS
    __builtin_amdgcn_global_load_async_to_lds_b128(
        (gas_v4i*)gsrc, (las_v4i*)ldst, 0, 0);
#else
    *(v8h*)ldst = *(const v8h*)gsrc;
#endif
}
__device__ __forceinline__ void async_wait0() {
#if HAS_ASYNC_LDS
    __builtin_amdgcn_s_wait_asynccnt(0);
#endif
}

__device__ __forceinline__ float sigmoid_f(float x) { return 1.f / (1.f + __expf(-x)); }
__device__ __forceinline__ float tanh_f(float x)    { return 2.f / (1.f + __expf(-2.f * x)) - 1.f; }

// ---------------------------------------------------------------------------
// f32 -> f16 convert with 2D zero-padding: out is [rd, cd], in is [rs, cs].
// ---------------------------------------------------------------------------
__global__ void cvt_pad_f16_k(const float* __restrict__ in, half_t* __restrict__ out,
                              int rs, int cs, int rd, int cd) {
    int i = blockIdx.x * blockDim.x + threadIdx.x;
    if (i >= rd * cd) return;
    int r = i / cd, c = i - r * cd;
    float v = (r < rs && c < cs) ? in[(size_t)r * cs + c] : 0.f;
    out[i] = (half_t)v;
}

// ---------------------------------------------------------------------------
// WMMA GEMM: out[m,n] = sum_k A[m,k]*Bt[n,k] + bias0[n] + bias1[n]
// Buffers pre-padded (rows to tile multiples, Kp multiple of 32) -> branch-free
// staging via async global->LDS DMA, double-buffered against the WMMA.
// block = 128 (4 waves); wave w owns N-tile blockIdx.y*4+w.
// ---------------------------------------------------------------------------
__global__ void wmma_gemm_bt_k(const half_t* __restrict__ A, int lda,
                               const half_t* __restrict__ Bt, int ldb,
                               const float* __restrict__ bias0,
                               const float* __restrict__ bias1,
                               float* __restrict__ out, int ldc,
                               int Mstore, int Nstore, int Kp) {
    __shared__ __align__(16) half_t lA[2][16 * 32];     // [buf][m][k]
    __shared__ __align__(16) half_t lB[2][4][16 * 32];  // [buf][wave][n][k]

    const int tid  = threadIdx.x;
    const int wave = tid >> 5;
    const int lane = tid & 31;
    const int hi   = lane >> 4;
    const int lr   = lane & 15;
    const int mBase = blockIdx.x * 16;
    const int nBase = (blockIdx.y * 4 + wave) * 16;

    // staging coordinates: A tile by threads 0..63 (1 chunk), B tile by each
    // wave's 32 lanes (2 chunks: rows br0 and br0+8).
    const int ar = tid >> 2, aq = tid & 3;
    const int br0 = lane >> 2, bq = lane & 3;

    float bv = 0.f;
    {
        int gn = nBase + lr;
        if (gn < Nstore) {
            if (bias0) bv += bias0[gn];
            if (bias1) bv += bias1[gn];
        }
    }
    v8f c;
    for (int i = 0; i < 8; ++i) c[i] = bv;

    const int kChunks = Kp >> 5;

    // prologue: stage chunk 0 into buffer 0
    if (tid < 64)
        cp16_g2l(&lA[0][ar * 32 + aq * 8], &A[(size_t)(mBase + ar) * lda + aq * 8]);
    cp16_g2l(&lB[0][wave][br0 * 32 + bq * 8],
             &Bt[(size_t)(nBase + br0) * ldb + bq * 8]);
    cp16_g2l(&lB[0][wave][(br0 + 8) * 32 + bq * 8],
             &Bt[(size_t)(nBase + br0 + 8) * ldb + bq * 8]);
    async_wait0();
    __syncthreads();

    for (int kc = 0; kc < kChunks; ++kc) {
        const int cb = kc & 1, nb = cb ^ 1;
        if (kc + 1 < kChunks) {           // stream next chunk while computing
            const int k1 = (kc + 1) << 5;
            if (tid < 64)
                cp16_g2l(&lA[nb][ar * 32 + aq * 8],
                         &A[(size_t)(mBase + ar) * lda + k1 + aq * 8]);
            cp16_g2l(&lB[nb][wave][br0 * 32 + bq * 8],
                     &Bt[(size_t)(nBase + br0) * ldb + k1 + bq * 8]);
            cp16_g2l(&lB[nb][wave][(br0 + 8) * 32 + bq * 8],
                     &Bt[(size_t)(nBase + br0 + 8) * ldb + k1 + bq * 8]);
        }
        HFrag a, b;
        a.h[0] = *(const v8h*)&lA[cb][lr * 32 + hi * 8];
        a.h[1] = *(const v8h*)&lA[cb][lr * 32 + 16 + hi * 8];
        b.h[0] = *(const v8h*)&lB[cb][wave][lr * 32 + hi * 8];
        b.h[1] = *(const v8h*)&lB[cb][wave][lr * 32 + 16 + hi * 8];
        c = __builtin_amdgcn_wmma_f32_16x16x32_f16(false, a.v, false, b.v,
                                                   (short)0, c, false, false);
        async_wait0();
        __syncthreads();
    }

    for (int i = 0; i < 8; ++i) {
        int gm = mBase + i + (hi ? 8 : 0);
        int gn = nBase + lr;
        if (gm < Mstore && gn < Nstore) out[(size_t)gm * ldc + gn] = c[i];
    }
}

// ---------------------------------------------------------------------------
// Persistent LSTM recurrence. One WG owns 16 batch rows for the whole T loop;
// w_hh, h (f16), c (f32), gate staging AND a double-buffered xg stage are all
// LDS-resident (236KB @ H=128). Step t+1's gates stream in via async DMA
// while step t computes -> recurrent critical path sees only LDS latency.
// block = 256 (8 waves); grid = 7; xg padded to 112 batch rows.
// ---------------------------------------------------------------------------
__global__ void lstm_persist_k(const float* __restrict__ xg,   // [112,T,4H]
                               const half_t* __restrict__ whh, // [4H,H]
                               half_t* __restrict__ hseq,      // [B,T,H] f16
                               int Bn, int T, int H) {
    extern __shared__ __align__(16) char smem[];
    const int fourH = 4 * H;
    half_t* whh_l = (half_t*)smem;                                        // [4H][H]
    float*  g_l   = (float*)(smem + (size_t)fourH * H * sizeof(half_t));  // [16][4H]
    float*  c_l   = g_l + 16 * fourH;                                     // [16][H]
    half_t* h_l   = (half_t*)(c_l + 16 * H);                              // [16][H]
    float*  xg_l  = (float*)(h_l + 16 * H);                               // [2][16][4H]

    const int tid  = threadIdx.x;
    const int wave = tid >> 5;
    const int lane = tid & 31;
    const int hi   = lane >> 4;
    const int lr   = lane & 15;
    const int mBase = blockIdx.x * 16;
    const int prow = tid >> 4;          // pointwise/staging row 0..15
    const int pcol = tid & 15;
    const int cpr  = fourH >> 2;        // 16-byte chunks per xg row

    // resident weights via async DMA + zero h,c state
    for (int idx = tid; idx < (fourH * H) >> 3; idx += 256)
        cp16_g2l(&whh_l[idx * 8], &whh[idx * 8]);
    for (int idx = tid; idx < 16 * H; idx += 256) {
        c_l[idx] = 0.f;
        h_l[idx] = (half_t)0;
    }
    // stage xg(t=0) into buffer 0
    {
        const float* srow = &xg[((size_t)(mBase + prow) * T + 0) * fourH];
        float* drow = &xg_l[prow * fourH];
        for (int j = pcol; j < cpr; j += 16) cp16_g2l(&drow[j * 4], &srow[j * 4]);
    }
    async_wait0();
    __syncthreads();

    const int tilesPerWave = fourH >> 7;   // (4H/16)/8 waves
    const int kChunks = H >> 5;

    for (int t = 0; t < T; ++t) {
        const int cb = t & 1, nb = cb ^ 1;
        if (t + 1 < T) {                  // stream next step's gates (async)
            const float* srow = &xg[((size_t)(mBase + prow) * T + t + 1) * fourH];
            float* drow = &xg_l[nb * 16 * fourH + prow * fourH];
            for (int j = pcol; j < cpr; j += 16) cp16_g2l(&drow[j * 4], &srow[j * 4]);
        }
        const float* xcur = &xg_l[cb * 16 * fourH];

        for (int nt = 0; nt < tilesPerWave; ++nt) {
            const int nBase = (wave * tilesPerWave + nt) * 16;
            const int gn = nBase + lr;
            v8f c;
            for (int i = 0; i < 8; ++i)
                c[i] = xcur[(i + (hi ? 8 : 0)) * fourH + gn];
            for (int kc = 0; kc < kChunks; ++kc) {
                const int k0 = kc << 5;
                HFrag a, b;
                a.h[0] = *(const v8h*)&h_l[lr * H + k0 + hi * 8];
                a.h[1] = *(const v8h*)&h_l[lr * H + k0 + 16 + hi * 8];
                b.h[0] = *(const v8h*)&whh_l[(size_t)gn * H + k0 + hi * 8];
                b.h[1] = *(const v8h*)&whh_l[(size_t)gn * H + k0 + 16 + hi * 8];
                c = __builtin_amdgcn_wmma_f32_16x16x32_f16(false, a.v, false, b.v,
                                                           (short)0, c, false, false);
            }
            for (int i = 0; i < 8; ++i)
                g_l[(i + (hi ? 8 : 0)) * fourH + gn] = c[i];
        }
        __syncthreads();

        // fused pointwise, gate order i,f,g,o; row = prow, cols strided by 16
        for (int hh = pcol; hh < H; hh += 16) {
            int idx = prow * H + hh;
            float si = sigmoid_f(g_l[prow * fourH + hh]);
            float sf = sigmoid_f(g_l[prow * fourH + H + hh]);
            float tg = tanh_f(g_l[prow * fourH + 2 * H + hh]);
            float so = sigmoid_f(g_l[prow * fourH + 3 * H + hh]);
            float cc = sf * c_l[idx] + si * tg;
            float hv = so * tanh_f(cc);
            c_l[idx] = cc;
            h_l[idx] = (half_t)hv;
            int gb = mBase + prow;
            if (gb < Bn) hseq[((size_t)gb * T + t) * H + hh] = (half_t)hv;
        }
        async_wait0();
        __syncthreads();
    }
}

// ---------------------------------------------------------------------------
// Host orchestration
// ---------------------------------------------------------------------------
static inline size_t alignUp(size_t v) { return (v + 255) & ~(size_t)255; }

extern "C" void kernel_launch(void* const* d_in, const int* in_sizes, int n_in,
                              void* d_out, int out_size, void* d_ws, size_t ws_size,
                              hipStream_t stream) {
    (void)in_sizes; (void)n_in; (void)out_size; (void)ws_size;
    const int B = 100, T = 75;
    const int Bp = 112;                  // padded batch (7 tiles of 16)
    const float* x = (const float*)d_in[0];

    struct Layer { int wih, whh, bih, bhh, I, H; };
    const Layer Ls[6] = {
        {1, 2, 3, 4, 75, 128}, {5, 6, 7, 8, 128, 128},
        {9, 10, 11, 12, 128, 64}, {13, 14, 15, 16, 64, 64},
        {17, 18, 19, 20, 64, 32}, {21, 22, 23, 24, 32, 32},
    };

    // workspace carve (16B-aligned, padded for branch-free staging)
    char* ws = (char*)d_ws;
    half_t* actA = (half_t*)ws; ws += alignUp((size_t)7504 * 128 * sizeof(half_t));
    half_t* actB = (half_t*)ws; ws += alignUp((size_t)7504 * 128 * sizeof(half_t));
    float*  xg   = (float*)ws;  ws += alignUp((size_t)Bp * T * 512 * sizeof(float));
    half_t* wihf = (half_t*)ws; ws += alignUp((size_t)512 * 128 * sizeof(half_t));
    half_t* whhf = (half_t*)ws; ws += alignUp((size_t)512 * 128 * sizeof(half_t));
    half_t* linf = (half_t*)ws; ws += alignUp((size_t)64 * 2400 * sizeof(half_t));

    // max dynamic LDS @ H=128: whh 128K + g 32K + c 8K + h 4K + xg 2x32K = 236KB
    int maxSmem = 4 * 128 * 128 * 2 + 16 * 4 * 128 * 4 + 16 * 128 * 4 + 16 * 128 * 2
                + 2 * 16 * 4 * 128 * 4;
    (void)hipFuncSetAttribute((const void*)lstm_persist_k,
                              hipFuncAttributeMaxDynamicSharedMemorySize, maxSmem);

    // x: [7500,75] f32 -> [7504,96] f16, zero-padded
    {
        int n = 7504 * 96;
        cvt_pad_f16_k<<<(n + 255) / 256, 256, 0, stream>>>(x, actA, 7500, 75, 7504, 96);
    }

    half_t* cur = actA;
    half_t* nxt = actB;
    int curLd = 96;                      // padded K of current activations
    for (int l = 0; l < 6; ++l) {
        const Layer& L = Ls[l];
        const int fourH = 4 * L.H;
        const int Kp = (l == 0) ? 96 : L.I;   // multiple of 32
        const float* wih = (const float*)d_in[L.wih];
        const float* whh = (const float*)d_in[L.whh];
        const float* bih = (const float*)d_in[L.bih];
        const float* bhh = (const float*)d_in[L.bhh];

        {   // w_ih -> f16, K padded
            int n = fourH * Kp;
            cvt_pad_f16_k<<<(n + 255) / 256, 256, 0, stream>>>(wih, wihf, fourH, L.I, fourH, Kp);
        }
        {   // w_hh -> f16
            int n = fourH * L.H;
            cvt_pad_f16_k<<<(n + 255) / 256, 256, 0, stream>>>(whh, whhf, fourH, L.H, fourH, L.H);
        }

        // xg = act * w_ih^T + b_ih + b_hh   (M = B*T; padded rows computed, not stored)
        {
            dim3 grid((B * T + 15) / 16, fourH / 64);
            wmma_gemm_bt_k<<<grid, 128, 0, stream>>>(cur, curLd, wihf, Kp, bih, bhh,
                                                     xg, fourH, B * T, fourH, Kp);
        }
        // persistent recurrence
        {
            size_t smem = (size_t)fourH * L.H * sizeof(half_t)
                        + (size_t)16 * fourH * sizeof(float)
                        + (size_t)16 * L.H * sizeof(float)
                        + (size_t)16 * L.H * sizeof(half_t)
                        + (size_t)2 * 16 * fourH * sizeof(float);
            lstm_persist_k<<<Bp / 16, 256, smem, stream>>>(xg, whhf, nxt, B, T, L.H);
        }
        half_t* tmp = cur; cur = nxt; nxt = tmp;
        curLd = L.H;
    }

    // final linear: out[100,60] = hseq[100,2400] * lin_w^T + lin_b
    {
        const float* lw = (const float*)d_in[25];
        const float* lb = (const float*)d_in[26];
        int n = 64 * 2400;   // rows 60..63 zero-padded -> branch-free B staging
        cvt_pad_f16_k<<<(n + 255) / 256, 256, 0, stream>>>(lw, linf, 60, 2400, 64, 2400);
        dim3 grid((100 + 15) / 16, 1);
        wmma_gemm_bt_k<<<grid, 128, 0, stream>>>(cur, 2400, linf, 2400, lb, nullptr,
                                                 (float*)d_out, 60, 100, 60, 2400);
    }
}
